// GATBaseline_82403242541170
// MI455X (gfx1250) — matrix operand order repaired and checked by reading, not compiled, not run on hardware
//
#include <hip/hip_runtime.h>
#include <hip/hip_bf16.h>

// ---------------- problem constants (match reference) ----------------
#define NN   50000      // nodes
#define EE   800000     // raw edges
#define ETOT (EE + NN)  // edges + self loops
#define DD   128        // embed dim
#define HH   128        // hidden dim
#define GG   64         // graphs

typedef __attribute__((ext_vector_type(16))) _Float16 v16h;
typedef __attribute__((ext_vector_type(4)))  _Float16 v4h;
typedef __attribute__((ext_vector_type(8)))  float    v8f;

static __device__ __forceinline__ void atomicMaxFloat(float* addr, float val) {
    // IEEE-754 ordered atomic max: ints compare like floats for >=0,
    // uints compare reversed for <0.  Works with -inf initialization.
    if (val >= 0.0f) {
        atomicMax((int*)addr, __float_as_int(val));
    } else {
        atomicMin((unsigned int*)addr, __float_as_uint(val));
    }
}

// ---------------- fills ----------------
__global__ void fill_k(float* __restrict__ p, float v, long n) {
    long i = (long)blockIdx.x * blockDim.x + threadIdx.x;
    if (i < n) p[i] = v;
}
__global__ void fill4_k(float4* __restrict__ p, float v, long n4) {
    long i = (long)blockIdx.x * blockDim.x + threadIdx.x;
    if (i < n4) p[i] = make_float4(v, v, v, v);
}

// ---------------- fused (gather +) GEMM + attention dots (WMMA) ----------------
// One block = 16 node rows x 128 cols.  8 waves, each owns a 16x16 tile.
// K = 128 done as 4 steps of v_wmma_f32_16x16x32_f16.
template <bool GATHER>
__global__ __launch_bounds__(256)
void gemm_node_k(const float* __restrict__ X,     // GATHER ? emb[V,128] : h_in[N,128]
                 const int*   __restrict__ xlex,  // node -> vocab row (GATHER only)
                 const float* __restrict__ W,     // [128,128] row-major (k,n)
                 const float* __restrict__ asrc,  // [128]
                 const float* __restrict__ adst,  // [128]
                 float* __restrict__ Hout,        // [N,128]
                 float* __restrict__ hs,          // [N]
                 float* __restrict__ hd)          // [N]
{
    constexpr int AS = 132;   // padded LDS strides (avoid bank conflicts)
    constexpr int WS = 132;
    __shared__ _Float16 Alds[16 * AS];     // A tile, f16
    __shared__ _Float16 Wlds[128 * WS];    // W transposed: [n][k], f16
    __shared__ float    Olds[16 * 128];    // f32 output tile

    const int t    = threadIdx.x;
    const int lane = t & 31;
    const int wave = t >> 5;
    const int row0 = blockIdx.x * 16;

    // ---- stage A tile with b128 loads (optionally embedding-gathered) ----
    #pragma unroll
    for (int i = 0; i < 2; ++i) {
        int idx = t + i * 256;              // 0..511 float4 slots
        int r = idx >> 5, c4 = (idx & 31) << 2;
        int g = row0 + r;
        long srow = GATHER ? (long)xlex[g] : (long)g;
        float4 v = ((const float4*)(X + srow * 128))[idx & 31];
        v4h h4 = { (_Float16)v.x, (_Float16)v.y, (_Float16)v.z, (_Float16)v.w };
        *(v4h*)&Alds[r * AS + c4] = h4;     // 8-byte aligned (AS even, c4 % 4 == 0)
    }
    // ---- stage W transposed with b128 loads (W stays hot in L2) ----
    #pragma unroll
    for (int i = 0; i < 16; ++i) {
        int idx = t + i * 256;              // 0..4095 float4 slots
        int k = idx >> 5, n4 = (idx & 31) << 2;
        float4 v = ((const float4*)W)[idx];
        Wlds[(n4 + 0) * WS + k] = (_Float16)v.x;
        Wlds[(n4 + 1) * WS + k] = (_Float16)v.y;
        Wlds[(n4 + 2) * WS + k] = (_Float16)v.z;
        Wlds[(n4 + 3) * WS + k] = (_Float16)v.w;
    }
    __syncthreads();

    const int colbase = wave * 16;
    const int m    = lane & 15;
    const int hsel = lane >> 4;

    v8f acc = {};
    #pragma unroll
    for (int kk = 0; kk < 4; ++kk) {
        const int kbase = kk * 32;
        v16h a, b;
        // A fragment: lane holds row m; K = hsel*16 + j
        #pragma unroll
        for (int j = 0; j < 16; ++j) {
            int k = kbase + hsel * 16 + j;
            a[j] = Alds[m * AS + k];
        }
        // B fragment: lane holds column (colbase+m); K = hsel*16 + j
        #pragma unroll
        for (int j = 0; j < 16; ++j) {
            int k = kbase + hsel * 16 + j;
            b[j] = Wlds[(colbase + m) * WS + k];
        }
        acc = __builtin_amdgcn_wmma_f32_16x16x32_f16(
                  false, a, false, b, (short)0, acc, false, false);
    }

    // ---- D tile -> LDS (C/D layout: lane: N=lane&15, M=8*hsel+v) ----
    #pragma unroll
    for (int v = 0; v < 8; ++v) {
        Olds[(hsel * 8 + v) * 128 + colbase + m] = acc[v];
    }
    __syncthreads();

    // ---- coalesced b128 global store ----
    #pragma unroll
    for (int i = 0; i < 2; ++i) {
        int idx = t + i * 256;              // 0..511 float4 slots
        int r = idx >> 5, c4 = (idx & 31) << 2;
        float4 v = *(const float4*)&Olds[r * 128 + c4];
        ((float4*)(Hout + (long)(row0 + r) * 128))[idx & 31] = v;
    }
    // ---- fused attention scalars hs = h.a_src, hd = h.a_dst ----
    if (t < 32) {
        int r = t & 15;
        const float* av = (t < 16) ? asrc : adst;
        float s = 0.0f;
        #pragma unroll 8
        for (int k = 0; k < 128; ++k) s += Olds[r * 128 + k] * av[k];
        if (t < 16) hs[row0 + r] = s; else hd[row0 + r] = s;
    }
}

// ---------------- edge phase ----------------
__global__ void edge_max_k(const int* __restrict__ ei,
                           const float* __restrict__ hs, const float* __restrict__ hd,
                           float* __restrict__ p, float* __restrict__ m)
{
    long e = (long)blockIdx.x * blockDim.x + threadIdx.x;
    if (e >= ETOT) return;
    int s, d;
    if (e < EE) { s = ei[e]; d = ei[EE + e]; } else { s = d = (int)(e - EE); }
    float x = hs[s] + hd[d];
    float lr = x > 0.0f ? x : 0.2f * x;      // leaky_relu(0.2)
    p[e] = lr;
    atomicMaxFloat(&m[d], lr);
}

__global__ void edge_exp_k(const int* __restrict__ ei,
                           const float* __restrict__ m,
                           float* __restrict__ p, float* __restrict__ z)
{
    long e = (long)blockIdx.x * blockDim.x + threadIdx.x;
    if (e >= ETOT) return;
    int d = (e < EE) ? ei[EE + e] : (int)(e - EE);
    float pe = __expf(p[e] - m[d]);
    p[e] = pe;
    atomicAdd(&z[d], pe);
}

// one wave per edge: 32 lanes x float4 = 128 cols
__global__ __launch_bounds__(256)
void edge_aggr_k(const int* __restrict__ ei,
                 const float* __restrict__ p, const float* __restrict__ z,
                 const float* __restrict__ Hin, float* __restrict__ Out)
{
    int lane = threadIdx.x & 31, wave = threadIdx.x >> 5;
    long e = (long)blockIdx.x * 8 + wave;
    if (e >= ETOT) return;
    int s, d;
    if (e < EE) { s = ei[e]; d = ei[EE + e]; } else { s = d = (int)(e - EE); }
    float alpha = p[e] / z[d];
    float4 v = ((const float4*)(Hin + (long)s * 128))[lane];
    float* o = Out + (long)d * 128 + lane * 4;
    atomicAdd(o + 0, alpha * v.x);
    atomicAdd(o + 1, alpha * v.y);
    atomicAdd(o + 2, alpha * v.z);
    atomicAdd(o + 3, alpha * v.w);
}

__global__ void bias_relu_k(float4* __restrict__ h, const float* __restrict__ b)
{
    long i4 = (long)blockIdx.x * blockDim.x + threadIdx.x;
    if (i4 >= (long)NN * 32) return;
    int c4 = (int)(i4 & 31) << 2;
    float4 v = h[i4];
    v.x = fmaxf(v.x + b[c4 + 0], 0.0f);
    v.y = fmaxf(v.y + b[c4 + 1], 0.0f);
    v.z = fmaxf(v.z + b[c4 + 2], 0.0f);
    v.w = fmaxf(v.w + b[c4 + 3], 0.0f);
    h[i4] = v;
}

// ---------------- pooling + head ----------------
__global__ void pool_k(const float4* __restrict__ h, const int* __restrict__ batch,
                       float* __restrict__ psum, float* __restrict__ pcnt)
{
    long i4 = (long)blockIdx.x * blockDim.x + threadIdx.x;
    if (i4 >= (long)NN * 32) return;
    int n = (int)(i4 >> 5), c4 = (int)(i4 & 31) << 2;
    int g = batch[n];
    float4 v = h[i4];
    float* o = psum + g * 128 + c4;
    atomicAdd(o + 0, v.x);
    atomicAdd(o + 1, v.y);
    atomicAdd(o + 2, v.z);
    atomicAdd(o + 3, v.w);
    if (c4 == 0) atomicAdd(&pcnt[g], 1.0f);
}

__global__ __launch_bounds__(128)
void final_k(const float* __restrict__ psum, const float* __restrict__ pcnt,
             const float* __restrict__ Wc, const float* __restrict__ bc,
             float* __restrict__ out)   // out[0:64]=logits, out[64:8256]=h_pool
{
    __shared__ float red[128];
    int g = blockIdx.x, t = threadIdx.x;
    float cnt = pcnt[g];
    float hp = psum[g * 128 + t] / fmaxf(cnt, 1.0f);
    out[GG + g * 128 + t] = hp;
    red[t] = hp * Wc[t];
    __syncthreads();
    for (int s = 64; s > 0; s >>= 1) {
        if (t < s) red[t] += red[t + s];
        __syncthreads();
    }
    if (t == 0) out[g] = red[0] + bc[0];
}

// ---------------- host launcher ----------------
static inline int cdiv(long a, long b) { return (int)((a + b - 1) / b); }

extern "C" void kernel_launch(void* const* d_in, const int* in_sizes, int n_in,
                              void* d_out, int out_size, void* d_ws, size_t ws_size,
                              hipStream_t stream)
{
    const int*   x_lex = (const int*)  d_in[0];
    const int*   eidx  = (const int*)  d_in[1];   // [2,E] row-major
    const int*   batch = (const int*)  d_in[2];
    const float* emb   = (const float*)d_in[3];
    const float* W1    = (const float*)d_in[4];
    const float* as1   = (const float*)d_in[5];
    const float* ad1   = (const float*)d_in[6];
    const float* b1    = (const float*)d_in[7];
    const float* W2    = (const float*)d_in[8];
    const float* as2   = (const float*)d_in[9];
    const float* ad2   = (const float*)d_in[10];
    const float* b2    = (const float*)d_in[11];
    const float* Wc    = (const float*)d_in[12];
    const float* bc    = (const float*)d_in[13];
    float* out = (float*)d_out;

    // workspace carve-up (floats): 2 ping-pong N x 128 buffers + scalars
    float* hA   = (float*)d_ws;          // [N,128] gemm output (h_pre)
    float* hB   = hA + (long)NN * 128;   // [N,128] agg output / layer input
    float* hs   = hB + (long)NN * 128;   // [N]
    float* hd   = hs + NN;               // [N]
    float* mbuf = hd + NN;               // [N]
    float* zbuf = mbuf + NN;             // [N]
    float* pbuf = zbuf + NN;             // [ETOT]
    float* psum = pbuf + ETOT;           // [G,128]
    float* pcnt = psum + GG * 128;       // [G]

    const long NH4 = (long)NN * 32;      // N*128/4 float4s
    const int  ETB = cdiv(ETOT, 256);

    // ---- init ----
    fill_k<<<cdiv(NN, 256), 256, 0, stream>>>(mbuf, -INFINITY, NN);
    fill_k<<<cdiv(NN, 256), 256, 0, stream>>>(zbuf, 0.0f, NN);
    fill4_k<<<cdiv(NH4, 256), 256, 0, stream>>>((float4*)hB, 0.0f, NH4);
    fill_k<<<cdiv(GG * 128 + GG, 256), 256, 0, stream>>>(psum, 0.0f, GG * 128 + GG);

    // ---- layer 1: gather(emb) @ W1 + att dots ----
    gemm_node_k<true><<<NN / 16, 256, 0, stream>>>(emb, x_lex, W1, as1, ad1, hA, hs, hd);
    edge_max_k<<<ETB, 256, 0, stream>>>(eidx, hs, hd, pbuf, mbuf);
    edge_exp_k<<<ETB, 256, 0, stream>>>(eidx, mbuf, pbuf, zbuf);
    edge_aggr_k<<<cdiv(ETOT, 8), 256, 0, stream>>>(eidx, pbuf, zbuf, hA, hB);
    bias_relu_k<<<cdiv(NH4, 256), 256, 0, stream>>>((float4*)hB, b1);

    // ---- re-init softmax state ----
    fill_k<<<cdiv(NN, 256), 256, 0, stream>>>(mbuf, -INFINITY, NN);
    fill_k<<<cdiv(NN, 256), 256, 0, stream>>>(zbuf, 0.0f, NN);

    // ---- layer 2: hB @ W2 -> hA ; then hB is dead, reuse as agg target ----
    gemm_node_k<false><<<NN / 16, 256, 0, stream>>>(hB, nullptr, W2, as2, ad2, hA, hs, hd);
    fill4_k<<<cdiv(NH4, 256), 256, 0, stream>>>((float4*)hB, 0.0f, NH4);
    edge_max_k<<<ETB, 256, 0, stream>>>(eidx, hs, hd, pbuf, mbuf);
    edge_exp_k<<<ETB, 256, 0, stream>>>(eidx, mbuf, pbuf, zbuf);
    edge_aggr_k<<<cdiv(ETOT, 8), 256, 0, stream>>>(eidx, pbuf, zbuf, hA, hB);
    bias_relu_k<<<cdiv(NH4, 256), 256, 0, stream>>>((float4*)hB, b2);

    // ---- mean pool + head ----
    pool_k<<<cdiv(NH4, 256), 256, 0, stream>>>((const float4*)hB, batch, psum, pcnt);
    final_k<<<GG, 128, 0, stream>>>(psum, pcnt, Wc, bc, out);

    (void)in_sizes; (void)n_in; (void)out_size; (void)ws_size;
}
